// LieConv_9783935500534
// MI455X (gfx1250) — matrix operand order repaired and checked by reading, not compiled, not run on hardware
//
#include <hip/hip_runtime.h>
#include <hip/hip_bf16.h>

typedef __attribute__((ext_vector_type(16))) _Float16 v16h;
typedef __attribute__((ext_vector_type(8)))  _Float16 half8;
typedef __attribute__((ext_vector_type(8)))  float    v8f;

#define BIGD 1e8f
#define NPTS 1024
#define BS   8
#define KNN  32
#define CCH  64
#define HID  32
#define PCH  16

__device__ __forceinline__ float swishf(float x) {
    return x / (1.0f + __expf(-x));
}

// ---------------------------------------------------------------------------
// K0: Wl (1024 x 64, f32, row-major [k][n]) -> WlT (64 x 1024, f16, [n][k])
// ---------------------------------------------------------------------------
__global__ void wl_transpose_kernel(const float* __restrict__ Wl,
                                    _Float16* __restrict__ WlT) {
    int t = blockIdx.x * blockDim.x + threadIdx.x;   // 0..65535
    if (t >= 64 * 1024) return;
    int n = t >> 10;          // 0..63
    int k = t & 1023;         // 0..1023
    WlT[t] = (_Float16)Wl[k * 64 + n];
}

// ---------------------------------------------------------------------------
// K1: per (b,m) row: dists over 1024 candidates, extract 32 smallest indices
//     (ascending distance, ties -> lower index, matching jax.lax.top_k(-d))
// ---------------------------------------------------------------------------
__global__ void topk_kernel(const float* __restrict__ abq,
                            const unsigned char* __restrict__ mask,
                            int* __restrict__ nidx) {
    __shared__ float sd[NPTS];
    const int bm = blockIdx.x;            // 0..8191
    const int b  = bm >> 10;
    const float* row = abq + (size_t)bm * NPTS * 3;

    for (int j = threadIdx.x; j < NPTS; j += blockDim.x) {
        float x = row[j * 3 + 0];
        float y = row[j * 3 + 1];
        float z = row[j * 3 + 2];
        float d = sqrtf(x * x + y * y + z * z);
        if (!mask[b * NPTS + j]) d = BIGD;
        sd[j] = d;
    }
    __syncthreads();

    if (threadIdx.x < 32) {
        const int lane = threadIdx.x;
        for (int r = 0; r < KNN; ++r) {
            float best = 3.0e38f;
            int   bidx = 1 << 30;
            #pragma unroll 4
            for (int i = 0; i < 32; ++i) {
                int   j = i * 32 + lane;
                float d = sd[j];
                if (d < best || (d == best && j < bidx)) { best = d; bidx = j; }
            }
            // wave32 butterfly reduce (value, index)
            #pragma unroll
            for (int off = 16; off > 0; off >>= 1) {
                float ov = __shfl_xor(best, off, 32);
                int   oi = __shfl_xor(bidx, off, 32);
                if (ov < best || (ov == best && oi < bidx)) { best = ov; bidx = oi; }
            }
            if (lane == 0) {
                nidx[bm * KNN + r] = bidx;
                sd[bidx] = BIGD;          // remove winner
            }
            // same-wave LDS ordering: make lane0's store visible before next scan
            asm volatile("s_wait_dscnt 0" ::: "memory");
            __builtin_amdgcn_wave_barrier();
        }
    }
}

// ---------------------------------------------------------------------------
// K2: per (b,m): gather V (f16, masked), MLP (parallelized 4 threads/neighbor)
//     -> WT (f16, masked), aggregation V^T(64x32) @ W(32x16) via
//     4x v_wmma_f32_16x16x32_f16. partial stored f16 row-major [bm][c*16+p]
// ---------------------------------------------------------------------------
__global__ void mlp_agg_kernel(const float* __restrict__ abq,
                               const float* __restrict__ vals,
                               const unsigned char* __restrict__ mask,
                               const float* __restrict__ W1, const float* __restrict__ b1,
                               const float* __restrict__ W2, const float* __restrict__ b2,
                               const float* __restrict__ W3, const float* __restrict__ b3,
                               const int* __restrict__ nidx,
                               _Float16* __restrict__ partial) {
    __shared__ float sW1[3 * HID];
    __shared__ float sB1[HID];
    __shared__ float sW2[HID * HID];
    __shared__ float sB2[HID];
    __shared__ float sW3[HID * PCH];
    __shared__ float sB3[PCH];
    __shared__ int   sIdx[KNN];
    __shared__ int   sVld[KNN];
    __shared__ float sH1[KNN][HID + 1];    // padded: odd stride -> conflict-free
    __shared__ float sH2[KNN][HID + 1];
    __shared__ _Float16 Vh[CCH * KNN];     // [c][k], k contiguous
    __shared__ _Float16 WT[PCH * KNN];     // [p][k], k contiguous

    const int tid = threadIdx.x;           // 0..127
    const int bm  = blockIdx.x;            // 0..8191
    const int b   = bm >> 10;
    const int m   = bm & 1023;

    for (int i = tid; i < 3 * HID;   i += 128) sW1[i] = W1[i];
    for (int i = tid; i < HID;       i += 128) sB1[i] = b1[i];
    for (int i = tid; i < HID * HID; i += 128) sW2[i] = W2[i];
    for (int i = tid; i < HID;       i += 128) sB2[i] = b2[i];
    for (int i = tid; i < HID * PCH; i += 128) sW3[i] = W3[i];
    for (int i = tid; i < PCH;       i += 128) sB3[i] = b3[i];
    if (tid < KNN) {
        int id = nidx[bm * KNN + tid];
        sIdx[tid] = id;
        sVld[tid] = (mask[b * NPTS + id] && mask[b * NPTS + m]) ? 1 : 0;
    }
    __syncthreads();

    const int k  = tid & 31;   // neighbor handled by this thread
    const int oq = tid >> 5;   // output-slice quarter (0..3)

    // gather neighbor values -> Vh[c][k] (f16, masked)
    for (int i = tid; i < CCH * KNN; i += 128) {
        int kk = i >> 6;         // 0..31
        int cc = i & 63;         // 0..63
        _Float16 v = (_Float16)0.0f;
        if (sVld[kk]) v = (_Float16)vals[((size_t)b * NPTS + sIdx[kk]) * CCH + cc];
        Vh[cc * KNN + kk] = v;
    }

    // --- MLP layer 1: 3 -> 32, each thread computes 8 outputs of neighbor k
    {
        const float* a = abq + ((size_t)bm * NPTS + sIdx[k]) * 3;
        float a0 = a[0], a1 = a[1], a2 = a[2];
        #pragma unroll
        for (int j = 0; j < 8; ++j) {
            int o = oq * 8 + j;
            float s = sB1[o] + a0 * sW1[o] + a1 * sW1[HID + o] + a2 * sW1[2 * HID + o];
            sH1[k][o] = swishf(s);
        }
    }
    __syncthreads();

    // --- MLP layer 2: 32 -> 32
    {
        float h1r[HID];
        #pragma unroll
        for (int i = 0; i < HID; ++i) h1r[i] = sH1[k][i];
        #pragma unroll
        for (int j = 0; j < 8; ++j) {
            int o = oq * 8 + j;
            float s = sB2[o];
            #pragma unroll
            for (int i = 0; i < HID; ++i) s += h1r[i] * sW2[i * HID + o];
            sH2[k][o] = swishf(s);
        }
    }
    __syncthreads();

    // --- MLP layer 3: 32 -> 16, masked, write B operand WT[p][k]
    {
        float h2r[HID];
        #pragma unroll
        for (int i = 0; i < HID; ++i) h2r[i] = sH2[k][i];
        const int vld = sVld[k];
        #pragma unroll
        for (int j = 0; j < 4; ++j) {
            int p = oq * 4 + j;
            float s = sB3[p];
            #pragma unroll
            for (int i = 0; i < HID; ++i) s += h2r[i] * sW3[i * PCH + p];
            float w = swishf(s);
            WT[p * KNN + k] = vld ? (_Float16)w : (_Float16)0.0f;
        }
    }
    __syncthreads();

    // aggregation: each wave computes one 16-row c-tile of (64x16) = V^T @ W
    const int wave  = tid >> 5;
    const int lane  = tid & 31;
    const int mloc  = lane & 15;
    const int abase = (lane < 16) ? 0 : 8;    // A 16-bit layout K-base
    const int kbase = (lane < 16) ? 0 : 16;   // unused; B is loaded with abase like A

    const int crow = wave * 16 + mloc;
    half8 alo = *(const half8*)&Vh[crow * KNN + abase];
    half8 ahi = *(const half8*)&Vh[crow * KNN + abase + 16];
    half8 blo = *(const half8*)&WT[mloc * KNN + abase];
    half8 bhi = *(const half8*)&WT[mloc * KNN + abase + 16];

    v16h A, B;
    #pragma unroll
    for (int j = 0; j < 8; ++j) {
        A[j] = alo[j]; A[j + 8] = ahi[j];
        B[j] = blo[j]; B[j + 8] = bhi[j];
    }
    v8f acc = {};
    acc = __builtin_amdgcn_wmma_f32_16x16x32_f16(false, A, false, B,
                                                 (short)0, acc, false, false);

    // store partial[bm][c*16 + p] as f16
    _Float16* prow = partial + (size_t)bm * (CCH * PCH);
    #pragma unroll
    for (int r = 0; r < 8; ++r) {
        int cr = wave * 16 + r + ((lane < 16) ? 0 : 8);
        prow[cr * PCH + mloc] = (_Float16)acc[r];
    }
}

// ---------------------------------------------------------------------------
// K3: final linear: (8192 x 1024) @ (1024 x 64) + bl, masked, f32 out
//     512 M-tile blocks x 4 waves (one 16-col N-tile each), K=1024 (32 WMMAs)
//     A tile (16 x 1024 f16 = 32KB) staged once per block into LDS via
//     global_load_async_to_lds_b128 (ASYNCcnt), rows padded to 2064B so the
//     16-lane fragment reads land on distinct bank groups.
// ---------------------------------------------------------------------------
#define AROW 1032   // 1024 halves + 8-half (16B) pad

__global__ void final_gemm_kernel(const _Float16* __restrict__ partial,
                                  const _Float16* __restrict__ WlT,
                                  const float* __restrict__ bl,
                                  const unsigned char* __restrict__ mask,
                                  float* __restrict__ out) {
    __shared__ _Float16 Ah[16 * AROW];    // ~33 KB

    const int tid   = threadIdx.x;        // 0..127
    const int wave  = tid >> 5;           // N-tile 0..3
    const int lane  = tid & 31;
    const int mloc  = lane & 15;
    const int abase = (lane < 16) ? 0 : 8;
    const int kbase = (lane < 16) ? 0 : 16;
    const int Mtile = blockIdx.x;         // 0..511

    // ---- async-stage A tile: 16 rows x 2048B -> LDS (padded rows) ----
    const unsigned ldsBase = (unsigned)(size_t)(&Ah[0]);
    const unsigned long long gbase =
        (unsigned long long)(const void*)(partial + (size_t)Mtile * 16 * 1024);
    for (int i = tid; i < 2048; i += 128) {          // 16B chunks
        int row = i >> 7;                            // 0..15
        int c16 = i & 127;                           // 0..127
        unsigned gv = (unsigned)(row * 2048 + c16 * 16);
        unsigned ld = ldsBase + (unsigned)(row * (AROW * 2) + c16 * 16);
        asm volatile("global_load_async_to_lds_b128 %0, %1, %2 offset:0"
                     :: "v"(ld), "v"(gv), "s"(gbase)
                     : "memory");
    }
    asm volatile("s_wait_asynccnt 0" ::: "memory");
    __syncthreads();

    const _Float16* arow = Ah  + (size_t)mloc * AROW;
    const _Float16* brow = WlT + (size_t)(wave * 16 + mloc) * 1024;

    v8f acc = {};
    for (int kk = 0; kk < 1024; kk += 32) {
        half8 alo = *(const half8*)(arow + kk + abase);
        half8 ahi = *(const half8*)(arow + kk + abase + 16);
        half8 blo = *(const half8*)(brow + kk + abase);
        half8 bhi = *(const half8*)(brow + kk + abase + 16);
        v16h A, B;
        #pragma unroll
        for (int j = 0; j < 8; ++j) {
            A[j] = alo[j]; A[j + 8] = ahi[j];
            B[j] = blo[j]; B[j + 8] = bhi[j];
        }
        acc = __builtin_amdgcn_wmma_f32_16x16x32_f16(false, A, false, B,
                                                     (short)0, acc, false, false);
    }

    #pragma unroll
    for (int r = 0; r < 8; ++r) {
        int mrow = r + ((lane < 16) ? 0 : 8);
        int bm   = Mtile * 16 + mrow;
        int nn   = wave * 16 + mloc;
        int b    = bm >> 10;
        int m    = bm & 1023;
        float v  = acc[r] + bl[nn];
        out[(size_t)bm * CCH + nn] = mask[b * NPTS + m] ? v : 0.0f;
    }
}

// ---------------------------------------------------------------------------
extern "C" void kernel_launch(void* const* d_in, const int* in_sizes, int n_in,
                              void* d_out, int out_size, void* d_ws, size_t ws_size,
                              hipStream_t stream) {
    const float*         abq  = (const float*)d_in[0];
    const float*         vals = (const float*)d_in[1];
    const unsigned char* mask = (const unsigned char*)d_in[2];
    const float* W1 = (const float*)d_in[3];
    const float* b1 = (const float*)d_in[4];
    const float* W2 = (const float*)d_in[5];
    const float* b2 = (const float*)d_in[6];
    const float* W3 = (const float*)d_in[7];
    const float* b3 = (const float*)d_in[8];
    const float* Wl = (const float*)d_in[9];
    const float* bl = (const float*)d_in[10];
    float* out = (float*)d_out;

    char* ws = (char*)d_ws;
    _Float16* WlT     = (_Float16*)ws;                             // 128 KB
    int*      nidx    = (int*)(ws + (128 << 10));                  //   1 MB
    _Float16* partial = (_Float16*)(ws + (128 << 10) + (1 << 20)); //  16 MB

    wl_transpose_kernel<<<256, 256, 0, stream>>>(Wl, WlT);
    topk_kernel<<<BS * NPTS, 256, 0, stream>>>(abq, mask, nidx);
    mlp_agg_kernel<<<BS * NPTS, 128, 0, stream>>>(abq, vals, mask,
                                                  W1, b1, W2, b2, W3, b3,
                                                  nidx, partial);
    final_gemm_kernel<<<(BS * NPTS) / 16, 128, 0, stream>>>(partial, WlT, bl, mask, out);
}
